// MambaBlock_10273561772152
// MI455X (gfx1250) — matrix operand order repaired and checked by reading, not compiled, not run on hardware
//
#include <hip/hip_runtime.h>

// ---------------------------------------------------------------------------
// Mamba block for MI455X (gfx1250, wave32, WMMA).
// Big GEMMs run on v_wmma_f32_16x16x32_bf16 (bf16 in, f32 accum),
// 32x64 output strip per wave (8 WMMAs / k-step, B-frags reused 2x).
// global_prefetch_b8 used to hide K-stream and scan-step latency.
// ---------------------------------------------------------------------------

#define D_MODEL 1024
#define D_STATE 16
#define D_CONV  4
#define D_INNER 2048
#define BATCH   4
#define SEQLEN  1024
#define NROWS   (BATCH * SEQLEN)          // 4096 token rows

typedef __attribute__((ext_vector_type(16))) __bf16 bf16x16;
typedef __attribute__((ext_vector_type(8)))  float  v8f;

union Frag {
    bf16x16 v;
    uint4   q[2];
};

// -------------------------- helpers ---------------------------------------

__device__ __forceinline__ unsigned short f2bf(float f) {
    unsigned int u = __float_as_uint(f);
    u += 0x7fffu + ((u >> 16) & 1u);   // round-to-nearest-even
    return (unsigned short)(u >> 16);
}

__device__ __forceinline__ float silu_f(float x) {
    return x / (1.0f + __expf(-x));
}

// -------------------------- pack f32 -> bf16 -------------------------------

__global__ __launch_bounds__(256)
void k_pack_bf16(const float* __restrict__ src, unsigned short* __restrict__ dst, int n) {
    int i = blockIdx.x * blockDim.x + threadIdx.x;
    if (i < n) dst[i] = f2bf(src[i]);
}

// -------------------------- WMMA GEMM --------------------------------------
// C[M,N](f32) = A[M,K](bf16, row-major) * W[N,K](bf16, row-major)^T
// 4 waves / block; each wave computes a 32(M) x 64(N) strip (8 accumulators).
// A/B fragments use contiguous K order: lane l holds K 16*(l/16) + 0..15 (same for A and B).

#define PF_ELTS 256   // prefetch distance along K (8 k-steps = 512 B/row)

__global__ __launch_bounds__(128)
void k_gemm_bf16_wmma(const unsigned short* __restrict__ A,
                      const unsigned short* __restrict__ W,
                      float* __restrict__ C,
                      int M, int N, int K) {
    const int lane = threadIdx.x & 31;
    const int wave = threadIdx.x >> 5;
    const int m0 = (blockIdx.y * 4 + wave) * 32;
    const int n0 = blockIdx.x * 64;

    const int am = lane & 15;              // A row within tile
    const int ak = (lane >> 4) << 4;       // lanes 16-31 hold K 16..31
    const int bn = lane & 15;              // B column within tile
    const int bk = (lane >> 4) << 4;       // lanes 16-31 hold K 16..31

    v8f acc_lo0 = {}, acc_lo1 = {}, acc_lo2 = {}, acc_lo3 = {};
    v8f acc_hi0 = {}, acc_hi1 = {}, acc_hi2 = {}, acc_hi3 = {};

    const unsigned short* Alo = A + (size_t)(m0 + am) * K;
    const unsigned short* Ahi = A + (size_t)(m0 + 16 + am) * K;
    const unsigned short* W0  = W + (size_t)(n0 +      bn) * K + bk;
    const unsigned short* W1  = W + (size_t)(n0 + 16 + bn) * K + bk;
    const unsigned short* W2  = W + (size_t)(n0 + 32 + bn) * K + bk;
    const unsigned short* W3  = W + (size_t)(n0 + 48 + bn) * K + bk;

    const uint4* AloQ = reinterpret_cast<const uint4*>(Alo);
    const uint4* AhiQ = reinterpret_cast<const uint4*>(Ahi);

    for (int k0 = 0; k0 < K; k0 += 32) {
        if ((k0 & (PF_ELTS - 1)) == 0 && k0 + PF_ELTS < K) {
            __builtin_prefetch(Alo + k0 + PF_ELTS, 0, 3);
            __builtin_prefetch(Ahi + k0 + PF_ELTS, 0, 3);
            __builtin_prefetch(W0  + k0 + PF_ELTS, 0, 3);
            __builtin_prefetch(W1  + k0 + PF_ELTS, 0, 3);
            __builtin_prefetch(W2  + k0 + PF_ELTS, 0, 3);
            __builtin_prefetch(W3  + k0 + PF_ELTS, 0, 3);
        }

        Frag alo, ahi;
        alo.q[0] = AloQ[(k0 + ak) >> 3];        // K = k0+ak   .. +7
        alo.q[1] = AloQ[(k0 + 8 + ak) >> 3];    // K = k0+8+ak .. +7
        ahi.q[0] = AhiQ[(k0 + ak) >> 3];
        ahi.q[1] = AhiQ[(k0 + 8 + ak) >> 3];

        Frag b0, b1, b2, b3;
        {
            const uint4* p = reinterpret_cast<const uint4*>(W0 + k0);
            b0.q[0] = p[0]; b0.q[1] = p[1];
        }
        {
            const uint4* p = reinterpret_cast<const uint4*>(W1 + k0);
            b1.q[0] = p[0]; b1.q[1] = p[1];
        }
        {
            const uint4* p = reinterpret_cast<const uint4*>(W2 + k0);
            b2.q[0] = p[0]; b2.q[1] = p[1];
        }
        {
            const uint4* p = reinterpret_cast<const uint4*>(W3 + k0);
            b3.q[0] = p[0]; b3.q[1] = p[1];
        }

        acc_lo0 = __builtin_amdgcn_wmma_f32_16x16x32_bf16(false, alo.v, false, b0.v, (short)0, acc_lo0, false, false);
        acc_hi0 = __builtin_amdgcn_wmma_f32_16x16x32_bf16(false, ahi.v, false, b0.v, (short)0, acc_hi0, false, false);
        acc_lo1 = __builtin_amdgcn_wmma_f32_16x16x32_bf16(false, alo.v, false, b1.v, (short)0, acc_lo1, false, false);
        acc_hi1 = __builtin_amdgcn_wmma_f32_16x16x32_bf16(false, ahi.v, false, b1.v, (short)0, acc_hi1, false, false);
        acc_lo2 = __builtin_amdgcn_wmma_f32_16x16x32_bf16(false, alo.v, false, b2.v, (short)0, acc_lo2, false, false);
        acc_hi2 = __builtin_amdgcn_wmma_f32_16x16x32_bf16(false, ahi.v, false, b2.v, (short)0, acc_hi2, false, false);
        acc_lo3 = __builtin_amdgcn_wmma_f32_16x16x32_bf16(false, alo.v, false, b3.v, (short)0, acc_lo3, false, false);
        acc_hi3 = __builtin_amdgcn_wmma_f32_16x16x32_bf16(false, ahi.v, false, b3.v, (short)0, acc_hi3, false, false);
    }

    // C/D layout: lanes 0-15 -> M = m0+r, lanes 16-31 -> M = m0+8+r
    const int cm = m0 + ((lane >> 4) << 3);
    const int cn = n0 + (lane & 15);
    float* Cp0 = C + (size_t)cm * N + cn;
    float* Cp1 = C + (size_t)(cm + 16) * N + cn;
#pragma unroll
    for (int r = 0; r < 8; ++r) {
        Cp0[(size_t)r * N +  0] = acc_lo0[r];
        Cp0[(size_t)r * N + 16] = acc_lo1[r];
        Cp0[(size_t)r * N + 32] = acc_lo2[r];
        Cp0[(size_t)r * N + 48] = acc_lo3[r];
        Cp1[(size_t)r * N +  0] = acc_hi0[r];
        Cp1[(size_t)r * N + 16] = acc_hi1[r];
        Cp1[(size_t)r * N + 32] = acc_hi2[r];
        Cp1[(size_t)r * N + 48] = acc_hi3[r];
    }
}

// -------------------- depthwise causal conv + SiLU -------------------------
// x_in[b,l,d] = silu( sum_j xz[b, l-3+j, d] * cw[d,0,j] + cb[d] )

__global__ __launch_bounds__(256)
void k_conv_silu(const float* __restrict__ XZ,   // [NROWS, 2*D_INNER], x part = cols [0,D_INNER)
                 const float* __restrict__ cw,   // [D_INNER, 1, D_CONV]
                 const float* __restrict__ cb,   // [D_INNER]
                 float* __restrict__ XIN) {      // [NROWS, D_INNER]
    int idx = blockIdx.x * blockDim.x + threadIdx.x;
    if (idx >= NROWS * D_INNER) return;
    int d   = idx & (D_INNER - 1);
    int row = idx >> 11;                     // D_INNER = 2048 = 2^11
    int l   = row & (SEQLEN - 1);

    float acc = cb[d];
#pragma unroll
    for (int j = 0; j < D_CONV; ++j) {
        int ll = l + j - (D_CONV - 1);
        if (ll >= 0)
            acc = fmaf(XZ[(size_t)(row + j - (D_CONV - 1)) * (2 * D_INNER) + d],
                       cw[d * D_CONV + j], acc);
    }
    XIN[(size_t)row * D_INNER + d] = silu_f(acc);
}

// -------------------- x_proj: [NROWS,2048] x [33,2048]^T -------------------

__global__ __launch_bounds__(256)
void k_xproj(const float* __restrict__ XIN, const float* __restrict__ Wp,
             float* __restrict__ XDBL) {
    __shared__ float sx[D_INNER];
    int row = blockIdx.x;
    for (int i = threadIdx.x; i < D_INNER; i += 256)
        sx[i] = XIN[(size_t)row * D_INNER + i];
    __syncthreads();
    int o = threadIdx.x;
    if (o < 2 * D_STATE + 1) {
        const float* w = Wp + (size_t)o * D_INNER;
        float acc = 0.f;
        for (int k = 0; k < D_INNER; ++k) acc = fmaf(sx[k], w[k], acc);
        XDBL[(size_t)row * (2 * D_STATE + 1) + o] = acc;
    }
}

// -------------------- selective scan (sequential over L) -------------------
// One block = 256 channels of one batch; thread keeps h[16] in registers.
// Latency-bound: prefetch next step's (dt,B,C) row and u while computing.

__global__ __launch_bounds__(256)
void k_scan(const float* __restrict__ XIN,
            const float* __restrict__ XDBL,
            const float* __restrict__ dtw,   // dt_proj_w [D_INNER,1]
            const float* __restrict__ dtb,   // dt_proj_b [D_INNER]
            const float* __restrict__ A_log, // [D_STATE]
            float* __restrict__ Y) {
    const int b     = blockIdx.x >> 3;            // 8 chunks per batch
    const int chunk = blockIdx.x & 7;
    const int ch    = chunk * 256 + threadIdx.x;

    float a[D_STATE], h[D_STATE];
#pragma unroll
    for (int s = 0; s < D_STATE; ++s) {
        a[s] = -__expf(fminf(A_log[s], 5.0f));
        h[s] = 0.0f;
    }
    const float wd = dtw[ch];
    const float bd = dtb[ch];

    __shared__ float sd[2 * D_STATE + 1];
    const float* xin_b = XIN + (size_t)b * SEQLEN * D_INNER + ch;
    const float* xd_b  = XDBL + (size_t)b * SEQLEN * (2 * D_STATE + 1);
    float*       y_b   = Y + (size_t)b * SEQLEN * D_INNER + ch;

    for (int t = 0; t < SEQLEN; ++t) {
        __syncthreads();
        if (threadIdx.x < 2 * D_STATE + 1)
            sd[threadIdx.x] = xd_b[(size_t)t * (2 * D_STATE + 1) + threadIdx.x];
        __syncthreads();

        float u = xin_b[(size_t)t * D_INNER];
        if (t + 1 < SEQLEN) {   // hide next step's L2 round-trip behind the VALU work
            __builtin_prefetch(xd_b + (size_t)(t + 1) * (2 * D_STATE + 1) + threadIdx.x, 0, 3);
            __builtin_prefetch(xin_b + (size_t)(t + 1) * D_INNER, 0, 3);
        }

        float dtv = fmaf(sd[0], wd, bd);
        float dt  = (dtv > 20.0f) ? dtv : log1pf(__expf(dtv));   // softplus
        dt = fminf(fmaxf(dt, 1e-4f), 10.0f);

        float y = 0.0f;
#pragma unroll
        for (int s = 0; s < D_STATE; ++s) {
            float dA  = __expf(fminf(fmaxf(dt * a[s], -20.0f), 0.0f));
            float dBu = dt * sd[1 + s] * u;
            dBu = fminf(fmaxf(dBu, -10.0f), 10.0f);
            float hn = fmaf(h[s], dA, dBu);
            h[s] = fminf(fmaxf(hn, -100.0f), 100.0f);
            y = fmaf(h[s], sd[1 + D_STATE + s], y);
        }
        y_b[(size_t)t * D_INNER] = y;
    }
}

// -------------------- layernorm + D*x_in + silu(z) gate, -> bf16 -----------

__global__ __launch_bounds__(256)
void k_ln_gate(const float* __restrict__ Y,
               const float* __restrict__ XIN,
               const float* __restrict__ XZ,     // z = cols [D_INNER, 2*D_INNER)
               const float* __restrict__ ln_g,
               const float* __restrict__ ln_b,
               const float* __restrict__ Dp,
               unsigned short* __restrict__ YGb) {
    __shared__ float rs[256], rq[256];
    const int row = blockIdx.x;
    const int tid = threadIdx.x;
    const float* y = Y + (size_t)row * D_INNER;

    float s1 = 0.f, s2 = 0.f;
    for (int i = tid; i < D_INNER; i += 256) {
        float v = y[i];
        s1 += v;
        s2 = fmaf(v, v, s2);
    }
    rs[tid] = s1; rq[tid] = s2;
    __syncthreads();
    for (int off = 128; off > 0; off >>= 1) {
        if (tid < off) { rs[tid] += rs[tid + off]; rq[tid] += rq[tid + off]; }
        __syncthreads();
    }
    const float mu   = rs[0] * (1.0f / D_INNER);
    const float var  = rq[0] * (1.0f / D_INNER) - mu * mu;
    const float rstd = rsqrtf(var + 1e-5f);

    for (int i = tid; i < D_INNER; i += 256) {
        float v = (y[i] - mu) * rstd * ln_g[i] + ln_b[i];
        v = fmaf(Dp[i], XIN[(size_t)row * D_INNER + i], v);
        float z = XZ[(size_t)row * (2 * D_INNER) + D_INNER + i];
        v *= silu_f(z);
        YGb[(size_t)row * D_INNER + i] = f2bf(v);
    }
}

// --------------------------- host launcher ---------------------------------
// inputs: 0 x, 1 in_proj_w, 2 conv_w, 3 conv_b, 4 x_proj_w, 5 dt_proj_w,
//         6 dt_proj_b, 7 A_log, 8 D, 9 out_proj_w, 10 ln_g, 11 ln_b

extern "C" void kernel_launch(void* const* d_in, const int* in_sizes, int n_in,
                              void* d_out, int out_size, void* d_ws, size_t ws_size,
                              hipStream_t stream) {
    (void)in_sizes; (void)n_in; (void)out_size; (void)ws_size;

    const float* x     = (const float*)d_in[0];
    const float* w_in  = (const float*)d_in[1];
    const float* cw    = (const float*)d_in[2];
    const float* cb    = (const float*)d_in[3];
    const float* w_xp  = (const float*)d_in[4];
    const float* dtw   = (const float*)d_in[5];
    const float* dtb   = (const float*)d_in[6];
    const float* A_log = (const float*)d_in[7];
    const float* Dp    = (const float*)d_in[8];
    const float* w_out = (const float*)d_in[9];
    const float* ln_g  = (const float*)d_in[10];
    const float* ln_b  = (const float*)d_in[11];
    float* out = (float*)d_out;

    // workspace layout (bytes, 256-aligned), total ~173 MB
    char* ws = (char*)d_ws;
    unsigned short* W1b  = (unsigned short*)(ws + 0);                     //  8 MB
    unsigned short* W2b  = (unsigned short*)(ws + 8388608);               //  4 MB
    unsigned short* Xb   = (unsigned short*)(ws + 12582912);              //  8 MB
    float*          XZ   = (float*)         (ws + 20971520);              // 64 MB
    float*          XIN  = (float*)         (ws + 88080384);              // 32 MB
    float*          XDBL = (float*)         (ws + 121634816);             // .5 MB
    float*          Y    = (float*)         (ws + 122175488);             // 32 MB
    unsigned short* YGb  = (unsigned short*)(ws + 155729920);             // 16 MB

    // 1) pack bf16 operands
    k_pack_bf16<<<(NROWS * D_MODEL + 255) / 256, 256, 0, stream>>>(x, Xb, NROWS * D_MODEL);
    k_pack_bf16<<<(2 * D_INNER * D_MODEL + 255) / 256, 256, 0, stream>>>(w_in, W1b, 2 * D_INNER * D_MODEL);
    k_pack_bf16<<<(D_MODEL * D_INNER + 255) / 256, 256, 0, stream>>>(w_out, W2b, D_MODEL * D_INNER);

    // 2) xz = x @ in_proj_w.T   (M=4096, N=4096, K=1024)
    k_gemm_bf16_wmma<<<dim3((2 * D_INNER) / 64, NROWS / 128), 128, 0, stream>>>(
        Xb, W1b, XZ, NROWS, 2 * D_INNER, D_MODEL);

    // 3) depthwise causal conv + SiLU
    k_conv_silu<<<(NROWS * D_INNER + 255) / 256, 256, 0, stream>>>(XZ, cw, cb, XIN);

    // 4) x_dbl = x_in @ x_proj_w.T
    k_xproj<<<NROWS, 256, 0, stream>>>(XIN, w_xp, XDBL);

    // 5) selective scan
    k_scan<<<BATCH * (D_INNER / 256), 256, 0, stream>>>(XIN, XDBL, dtw, dtb, A_log, Y);

    // 6) layernorm + residual D*x_in + silu(z) gate -> bf16
    k_ln_gate<<<NROWS, 256, 0, stream>>>(Y, XIN, XZ, ln_g, ln_b, Dp, YGb);

    // 7) out = y_gated @ out_proj_w.T   (M=4096, N=1024, K=2048)
    k_gemm_bf16_wmma<<<dim3(D_MODEL / 64, NROWS / 128), 128, 0, stream>>>(
        YGb, W2b, out, NROWS, D_MODEL, D_INNER);
}
